// G_GRU_net_73821897883786
// MI455X (gfx1250) — matrix operand order repaired and checked by reading, not compiled, not run on hardware
//
#include <hip/hip_runtime.h>
#include <math.h>

#define NN 15
#define BB 16
#define SS 256
#define HH 256
#define FINx 12
#define FOUTx 12
#define G3 768           // 3*H (r|u|c)
#define NTILES 48        // G3/16

typedef __attribute__((ext_vector_type(16))) __bf16 v16bf;
typedef __attribute__((ext_vector_type(8)))  float  v8f;

// ---------------------------------------------------------------------------
// 16-bit A-matrix 16x32 fragment map (ISA 7.12.2): (m,k)->(lane, elem idx)
// VGPR v (v<4): k = 2v + 8*hi + j ; (v>=4): k = 16 + 2(v-4) + 8*hi + j
// lane = m + 16*hi ; elem idx = 2v + j
// ---------------------------------------------------------------------------
__device__ __forceinline__ void amap(int m, int k, int& lane, int& idx) {
  int hi, r, v;
  if (k < 16) { hi = k >> 3; r = k & 7; v = r >> 1; }
  else        { int kp = k - 16; hi = kp >> 3; r = kp & 7; v = 4 + (r >> 1); }
  lane = m + (hi << 4);
  idx  = (v << 1) + (k & 1);
}

__device__ __forceinline__ v8f wmma_bf16(v16bf a, v16bf b, v8f c) {
  return __builtin_amdgcn_wmma_f32_16x16x32_bf16(false, a, false, b, (short)0, c, false, false);
}
__device__ __forceinline__ v8f vzero8() {
  v8f z;
#pragma unroll
  for (int i = 0; i < 8; ++i) z[i] = 0.f;
  return z;
}
__device__ __forceinline__ float sigmf(float x) { return 1.f / (1.f + expf(-x)); }

// ---------------------------------------------------------------------------
// Weight prep: V = U @ gcn, packed into B-fragment layout [nt][kt][lane(=n%16+16h)][i], k%32 = 16(i/8)+8h+i%8
// ---------------------------------------------------------------------------
struct VArgsT { const float* U[12]; const float* G[4]; };

__global__ __launch_bounds__(256) void prep_V(VArgsT a, __bf16* __restrict__ dst) {
  int gid = blockIdx.x * 256 + threadIdx.x;            // 4 * 48 * 8 * 512
  if (gid >= 4 * 196608) return;
  int cell = gid / 196608;        int rem  = gid - cell * 196608;
  int nt   = rem / 4096;          int rem2 = rem - nt * 4096;
  int kt   = rem2 >> 9;           int le   = rem2 & 511;
  int lane = le >> 4;             int i    = le & 15;
  int k = (kt << 5) + ((i >> 3) << 4) + ((lane >> 4) << 3) + (i & 7);
  int nc = (nt << 4) + (lane & 15);
  int g = nc >> 8, row = nc & 255;
  const float* U  = a.U[cell * 3 + g];
  const float* Gm = a.G[cell];
  float s = 0.f;
  for (int j = 0; j < HH; ++j) s += U[row * HH + j] * Gm[j * HH + k];
  dst[gid] = (__bf16)s;
}

struct WArgsT { const float* W[6]; };

__global__ __launch_bounds__(256) void prep_W(WArgsT a, int kin, __bf16* __restrict__ dst) {
  int ktx = kin >> 5;
  int percell = NTILES * ktx * 512;
  int gid = blockIdx.x * 256 + threadIdx.x;
  if (gid >= 2 * percell) return;
  int cell = gid / percell;       int rem  = gid - cell * percell;
  int nt   = rem / (ktx * 512);   int rem2 = rem - nt * (ktx * 512);
  int kt   = rem2 >> 9;           int le   = rem2 & 511;
  int lane = le >> 4;             int i    = le & 15;
  int k = (kt << 5) + ((i >> 3) << 4) + ((lane >> 4) << 3) + (i & 7);
  int nc = (nt << 4) + (lane & 15);
  int g = nc >> 8, row = nc & 255;
  const float* W = a.W[cell * 3 + g];
  dst[gid] = (__bf16)W[row * kin + k];
}

__global__ __launch_bounds__(256) void prep_kout(const float* __restrict__ ko, __bf16* __restrict__ dst) {
  int gid = blockIdx.x * 256 + threadIdx.x;            // 16 * 512
  if (gid >= 16 * 512) return;
  int kt = gid >> 9; int le = gid & 511;
  int lane = le >> 4; int i = le & 15;
  int k = (kt << 5) + ((i >> 3) << 4) + ((lane >> 4) << 3) + (i & 7);
  int n = lane & 15;
  dst[gid] = (__bf16)((n < FOUTx) ? ko[n * (2 * HH) + k] : 0.f);
}

// ---------------------------------------------------------------------------
// lin_in: node-mix then (x @ k.T + b, relu), output in A-fragment layout (K=256)
// ---------------------------------------------------------------------------
__global__ __launch_bounds__(256) void nodemix_in(const float* __restrict__ x,
                                                  const float* __restrict__ adjI,
                                                  float* __restrict__ xm) {
  int gid = blockIdx.x * 256 + threadIdx.x;            // 4096*15*12
  if (gid >= BB * SS * NN * FINx) return;
  int bs = gid / (NN * FINx); int r = gid - bs * (NN * FINx);
  int m = r / FINx, f = r - m * FINx;
  float s = 0.f;
  for (int n = 0; n < NN; ++n) s += adjI[m * NN + n] * x[bs * (NN * FINx) + n * FINx + f];
  xm[gid] = s;
}

__global__ __launch_bounds__(256) void lin_in(const float* __restrict__ xm,
                                              const float* __restrict__ kk,
                                              const float* __restrict__ bb,
                                              __bf16* __restrict__ y0A) {
  int gid = blockIdx.x * 256 + threadIdx.x;            // 4096*15*256
  if (gid >= BB * SS * NN * HH) return;
  int bs = gid / (NN * HH); int r = gid - bs * (NN * HH);
  int m = r >> 8, hc = r & 255;
  float s = bb[hc];
  for (int f = 0; f < FINx; ++f) s += xm[bs * (NN * FINx) + m * FINx + f] * kk[hc * FINx + f];
  s = fmaxf(s, 0.f);
  int kt = hc >> 5, kr = hc & 31, lane, i;
  amap(m, kr, lane, i);
  y0A[((size_t)bs * 8 + kt) * 512 + lane * 16 + i] = (__bf16)s;
}

// ---------------------------------------------------------------------------
// Persistent scan: one workgroup per (batch, dir); h kept in LDS across 256 steps.
// Per step: Zx = X_t @ Wcat.T ; Zh = h @ Vcat.T (both WMMA, A-frags hoisted to
// registers once per step) ; msg = NodeMix(Zh); gates; h update; write concat
// output in A-frag layout (K=512).
// ---------------------------------------------------------------------------
template <int KTX>
__global__ __launch_bounds__(256) void scan_layer(
    const __bf16* __restrict__ Xfrag,
    const __bf16* __restrict__ Wp0, const __bf16* __restrict__ Wp1,
    const __bf16* __restrict__ Vp0, const __bf16* __restrict__ Vp1,
    const float* br0, const float* bu0, const float* bc0,
    const float* br1, const float* bu1, const float* bc1,
    const float* adj0, const float* adj1,
    const float* hin0, const float* hin1, int use_init,
    float* __restrict__ hfin0, float* __restrict__ hfin1,
    __bf16* __restrict__ Yfrag) {
  const int b   = blockIdx.x;
  const int dir = blockIdx.y;
  const int tid = threadIdx.x;
  const int wv = tid >> 5, lane = tid & 31;

  const __bf16* Wp  = dir ? Wp1 : Wp0;
  const __bf16* Vp  = dir ? Vp1 : Vp0;
  const float*  brp = dir ? br1 : br0;
  const float*  bup = dir ? bu1 : bu0;
  const float*  bcp = dir ? bc1 : bc0;
  const float*  adjp = dir ? adj1 : adj0;
  const float*  hinp = dir ? hin1 : hin0;
  float*        hfin = dir ? hfin1 : hfin0;

  extern __shared__ unsigned char smem[];
  float*  zh    = (float*)smem;                 // 16*768
  float*  zx    = zh + 16 * G3;                 // 16*768
  __bf16* hfrag = (__bf16*)(zx + 16 * G3);      // 8*32*16 (A-frag layout, 32B aligned)
  float*  h32   = (float*)(hfrag + 4096);       // 15*256
  float*  adj_s = h32 + NN * HH;                // 240 (padded)
  float*  bias_s = adj_s + 240;                 // 768

  for (int e = tid; e < NN * NN; e += 256) adj_s[e] = adjp[e];
  for (int e = tid; e < G3; e += 256) {
    float v;
    if (e < 256) v = brp[e]; else if (e < 512) v = bup[e - 256]; else v = bcp[e - 512];
    bias_s[e] = v;
  }
  // init h (f32 + bf16 fragment copy); zero fragment row m=15
  for (int e = tid; e < NN * HH; e += 256) {
    int n = e >> 8, hc = e & 255;
    float h = use_init ? hinp[((size_t)b * NN + n) * HH + hc] : 0.f;
    h32[e] = h;
    int kt = hc >> 5, kr = hc & 31, la, ii;
    amap(n, kr, la, ii);
    hfrag[kt * 512 + la * 16 + ii] = (__bf16)h;
  }
  {
    int k = tid;               // tid covers 0..255 exactly
    int kt = k >> 5, kr = k & 31, la, ii;
    amap(15, kr, la, ii);
    hfrag[kt * 512 + la * 16 + ii] = (__bf16)0.f;
  }
  __syncthreads();

  for (int ti = 0; ti < SS; ++ti) {
    const int t = dir ? (SS - 1 - ti) : ti;
    const __bf16* Xb = Xfrag + (size_t)(b * SS + t) * KTX * 512;

    // --- hoist A fragments for this step into registers ---
    v16bf ax[KTX];
#pragma unroll
    for (int kt = 0; kt < KTX; ++kt)
      ax[kt] = *(const v16bf*)(Xb + (size_t)kt * 512 + lane * 16);
    v16bf ah[8];
#pragma unroll
    for (int kt = 0; kt < 8; ++kt)
      ah[kt] = *(const v16bf*)(hfrag + kt * 512 + lane * 16);

    // --- WMMA GEMMs: 48 N-tiles split over 8 waves ---
    for (int q = 0; q < 6; ++q) {
      const int nt = wv * 6 + q;
      v8f accx = vzero8(), acch = vzero8();
#pragma unroll
      for (int kt = 0; kt < KTX; ++kt) {
        v16bf w = *(const v16bf*)(Wp + ((size_t)nt * KTX + kt) * 512 + lane * 16);
        accx = wmma_bf16(ax[kt], w, accx);
      }
#pragma unroll
      for (int kt = 0; kt < 8; ++kt) {
        v16bf w = *(const v16bf*)(Vp + ((size_t)nt * 8 + kt) * 512 + lane * 16);
        acch = wmma_bf16(ah[kt], w, acch);
      }
#pragma unroll
      for (int v = 0; v < 8; ++v) {            // C/D layout: m = v + 8*(lane/16), n = lane%16
        int m = v + ((lane >> 4) << 3);
        int col = nt * 16 + (lane & 15);
        zx[m * G3 + col] = accx[v];
        zh[m * G3 + col] = acch[v];
      }
    }
    __syncthreads();

    // --- node mix + gates + h update + output write ---
    for (int e = tid; e < NN * HH; e += 256) {
      int n = e >> 8, hc = e & 255;
      float mr = 0.f, mu = 0.f, mc = 0.f;
      for (int m = 0; m < NN; ++m) {
        float a = adj_s[m * NN + n];           // msg[n] = sum_m adj[m][n] * Zh[m]
        mr += a * zh[m * G3 + hc];
        mu += a * zh[m * G3 + 256 + hc];
        mc += a * zh[m * G3 + 512 + hc];
      }
      float r = sigmf(zx[n * G3 + hc]       + bias_s[hc]       + mr);
      float u = sigmf(zx[n * G3 + 256 + hc] + bias_s[256 + hc] + mu);
      float c = tanhf(zx[n * G3 + 512 + hc] + bias_s[512 + hc] + r * mc);
      float hn = u * h32[e] + (1.f - u) * c;
      h32[e] = hn;
      int kt = hc >> 5, kr = hc & 31, la, ii;
      amap(n, kr, la, ii);
      hfrag[kt * 512 + la * 16 + ii] = (__bf16)hn;
      int ko = dir * HH + hc;                  // concat [yf|yb] along feature axis
      int kt2 = ko >> 5, kr2 = ko & 31;
      amap(n, kr2, la, ii);
      Yfrag[((size_t)(b * SS + t) * 16 + kt2) * 512 + la * 16 + ii] = (__bf16)hn;
    }
    __syncthreads();
  }

  for (int e = tid; e < NN * HH; e += 256) {
    int n = e >> 8, hc = e & 255;
    hfin[((size_t)b * NN + n) * HH + hc] = h32[e];
  }
}

// ---------------------------------------------------------------------------
// lin_out: per (b,s), WMMA (15x512)@(512x16), then node-mix + bias
// ---------------------------------------------------------------------------
__global__ __launch_bounds__(256) void lin_out_kernel(const __bf16* __restrict__ Y2,
                                                      const __bf16* __restrict__ Kp,
                                                      const float* __restrict__ adjO,
                                                      const float* __restrict__ bO,
                                                      float* __restrict__ out) {
  const int tid = threadIdx.x;
  const int wv = tid >> 5, lane = tid & 31;
  const int bs = blockIdx.x * 8 + wv;          // 512 blocks * 8 waves = 4096

  __shared__ float zz[8][256];
  v8f acc = vzero8();
#pragma unroll
  for (int kt = 0; kt < 16; ++kt) {
    v16bf a = *(const v16bf*)(Y2 + ((size_t)bs * 16 + kt) * 512 + lane * 16);
    v16bf w = *(const v16bf*)(Kp + (size_t)kt * 512 + lane * 16);
    acc = wmma_bf16(a, w, acc);
  }
#pragma unroll
  for (int v = 0; v < 8; ++v) {
    int m = v + ((lane >> 4) << 3);
    int col = lane & 15;
    zz[wv][m * 16 + col] = acc[v];
  }
  __syncthreads();
  for (int e = lane; e < NN * FOUTx; e += 32) {
    int m = e / FOUTx, o = e - m * FOUTx;
    float s = bO[o];
    for (int n = 0; n < NN; ++n) s += adjO[m * NN + n] * zz[wv][n * 16 + o];
    out[(size_t)bs * (NN * FOUTx) + m * FOUTx + o] = s;
  }
}

// ---------------------------------------------------------------------------
// Host side
// ---------------------------------------------------------------------------
struct HostCell { const float *Wr,*br,*Wu,*bu,*Wc,*bc,*Ur,*Uu,*Uc,*adj,*gcn; };

static HostCell read_cell_insertion(void* const* d, int base) {
  HostCell c;
  c.Wr=(const float*)d[base+0];  c.br=(const float*)d[base+1];
  c.Wu=(const float*)d[base+2];  c.bu=(const float*)d[base+3];
  c.Wc=(const float*)d[base+4];  c.bc=(const float*)d[base+5];
  c.Ur=(const float*)d[base+6];  c.Uu=(const float*)d[base+7];
  c.Uc=(const float*)d[base+8];  c.adj=(const float*)d[base+9];
  c.gcn=(const float*)d[base+10];
  return c;
}
static HostCell read_cell_sorted(void* const* d, int base) {
  // sorted keys: Uc,Ur,Uu,Wc,Wr,Wu,adj,bc,br,bu,gcn
  HostCell c;
  c.Uc=(const float*)d[base+0];  c.Ur=(const float*)d[base+1];
  c.Uu=(const float*)d[base+2];  c.Wc=(const float*)d[base+3];
  c.Wr=(const float*)d[base+4];  c.Wu=(const float*)d[base+5];
  c.adj=(const float*)d[base+6]; c.bc=(const float*)d[base+7];
  c.br=(const float*)d[base+8];  c.bu=(const float*)d[base+9];
  c.gcn=(const float*)d[base+10];
  return c;
}

extern "C" void kernel_launch(void* const* d_in, const int* in_sizes, int n_in,
                              void* d_out, int out_size, void* d_ws, size_t ws_size,
                              hipStream_t stream) {
  (void)n_in; (void)out_size; (void)ws_size;
  const float *x, *li_k, *li_adj, *li_b, *lo_k, *lo_adj, *lo_b;
  HostCell C[4];   // rnn1.fwd, rnn1.bwd, rnn2.fwd, rnn2.bwd

  if (in_sizes[0] == BB * SS * NN * FINx) {   // insertion-order flatten: x first
    x = (const float*)d_in[0];
    li_k = (const float*)d_in[1]; li_adj = (const float*)d_in[2]; li_b = (const float*)d_in[3];
    C[0] = read_cell_insertion(d_in, 4);
    C[1] = read_cell_insertion(d_in, 15);
    C[2] = read_cell_insertion(d_in, 26);
    C[3] = read_cell_insertion(d_in, 37);
    lo_k = (const float*)d_in[48]; lo_adj = (const float*)d_in[49]; lo_b = (const float*)d_in[50];
  } else {                                    // jax pytree sorted-key flatten: params first
    li_adj = (const float*)d_in[0]; li_b = (const float*)d_in[1]; li_k = (const float*)d_in[2];
    lo_adj = (const float*)d_in[3]; lo_b = (const float*)d_in[4]; lo_k = (const float*)d_in[5];
    C[1] = read_cell_sorted(d_in, 6);   // rnn1.bwd
    C[0] = read_cell_sorted(d_in, 17);  // rnn1.fwd
    C[3] = read_cell_sorted(d_in, 28);  // rnn2.bwd
    C[2] = read_cell_sorted(d_in, 39);  // rnn2.fwd
    x = (const float*)d_in[50];
  }

  // ---- workspace carve ----
  size_t off = 0;
  auto carve = [&](size_t bytes) -> void* {
    void* p = (char*)d_ws + off;
    off = (off + bytes + 255) & ~(size_t)255;
    return p;
  };
  float*  xm    = (float*)carve((size_t)BB * SS * NN * FINx * 4);
  __bf16* y0A   = (__bf16*)carve((size_t)BB * SS * 8 * 512 * 2);
  __bf16* y1A   = (__bf16*)carve((size_t)BB * SS * 16 * 512 * 2);
  __bf16* y2A   = (__bf16*)carve((size_t)BB * SS * 16 * 512 * 2);
  __bf16* vpack = (__bf16*)carve((size_t)4 * 196608 * 2);
  __bf16* wpk1  = (__bf16*)carve((size_t)2 * 196608 * 2);  // layer1: K=256
  __bf16* wpk2  = (__bf16*)carve((size_t)2 * 393216 * 2);  // layer2: K=512
  __bf16* kpk   = (__bf16*)carve((size_t)16 * 512 * 2);
  float*  fin_f = (float*)carve((size_t)BB * NN * HH * 4);
  float*  fin_b = (float*)carve((size_t)BB * NN * HH * 4);

  // ---- weight prep ----
  VArgsT va;
  for (int c = 0; c < 4; ++c) {
    va.U[c * 3 + 0] = C[c].Ur; va.U[c * 3 + 1] = C[c].Uu; va.U[c * 3 + 2] = C[c].Uc;
    va.G[c] = C[c].gcn;
  }
  prep_V<<<3072, 256, 0, stream>>>(va, vpack);

  WArgsT w1 = {{C[0].Wr, C[0].Wu, C[0].Wc, C[1].Wr, C[1].Wu, C[1].Wc}};
  WArgsT w2 = {{C[2].Wr, C[2].Wu, C[2].Wc, C[3].Wr, C[3].Wu, C[3].Wc}};
  prep_W<<<1536, 256, 0, stream>>>(w1, 256, wpk1);
  prep_W<<<3072, 256, 0, stream>>>(w2, 512, wpk2);
  prep_kout<<<32, 256, 0, stream>>>(lo_k, kpk);

  // ---- lin_in ----
  nodemix_in<<<2880, 256, 0, stream>>>(x, li_adj, xm);
  lin_in<<<61440, 256, 0, stream>>>(xm, li_k, li_b, y0A);

  // ---- scans ----
  const size_t SMEM = (size_t)(16 * G3 * 2 + NN * HH + 240 + G3) * 4 + 4096 * 2; // 125,888 B
  scan_layer<8><<<dim3(BB, 2), 256, SMEM, stream>>>(
      y0A,
      wpk1, wpk1 + 196608, vpack, vpack + 196608,
      C[0].br, C[0].bu, C[0].bc, C[1].br, C[1].bu, C[1].bc,
      C[0].adj, C[1].adj,
      nullptr, nullptr, 0,
      fin_f, fin_b, y1A);

  float* out = (float*)d_out;
  scan_layer<16><<<dim3(BB, 2), 256, SMEM, stream>>>(
      y1A,
      wpk2, wpk2 + 393216, vpack + 2 * 196608, vpack + 3 * 196608,
      C[2].br, C[2].bu, C[2].bc, C[3].br, C[3].bu, C[3].bc,
      C[2].adj, C[3].adj,
      fin_f, fin_b, 1,
      out + BB * SS * NN * FOUTx,
      out + BB * SS * NN * FOUTx + BB * NN * HH,
      y2A);

  // ---- lin_out ----
  lin_out_kernel<<<512, 256, 0, stream>>>(y2A, kpk, lo_adj, lo_b, out);
}
